// ZeroShotSegmentAnything_61641370632332
// MI455X (gfx1250) — compile-verified, matrix operand order checked by reading, not run on hardware
//
#include <hip/hip_runtime.h>
#include <hip/hip_bf16.h>
#include <math.h>

typedef __attribute__((ext_vector_type(16))) _Float16 v16h;
typedef __attribute__((ext_vector_type(8)))  float    v8f;

#define EMBED_C 256
#define NPIX    4096      // 64*64 feature pixels
#define FEAT    64
#define PH      576       // cropped intermediate height (2160 * 1024/3840 rounded)
#define PW      1024      // cropped intermediate width
#define H_OUT   2160
#define W_OUT   3840
#define CELL    240       // 240x240 output pixels per grid cell
#define NCS     16
#define NCELLS  256
#define THRESH  0.65f

// ---- workspace layout (bytes) ----
#define WS_SIM_OFF   0                        // float[4096]
#define WS_U_OFF     16384                    // float[576*1024]
#define WS_CELL_OFF  (16384 + PH*PW*4)        // u64[256]  (8-aligned)
#define WS_GMIN_OFF  (WS_CELL_OFF + NCELLS*8) // u64[1]

#if __has_builtin(__builtin_amdgcn_global_load_async_to_lds_b128)
#define HAVE_ASYNC_LDS 1
// Types matching the builtin's parameter types (from the clang diagnostic):
// param 1 = global pointer to int __attribute__((vector_size(16)))
typedef int zs_v4i __attribute__((vector_size(16)));
typedef __attribute__((address_space(1))) zs_v4i* zs_gv4p;
typedef __attribute__((address_space(3))) zs_v4i* zs_lv4p;
#else
#define HAVE_ASYNC_LDS 0
#endif

// ------------------------------------------------------------------
// Kernel 0: re-init reduction slots (ws is poisoned; must reset per call)
// ------------------------------------------------------------------
__global__ void zssam_init_kernel(unsigned long long* __restrict__ cells,
                                  unsigned long long* __restrict__ gmin) {
    int t = threadIdx.x;
    if (t < NCELLS) cells[t] = 0ull;
    if (t == 0)     gmin[0]  = ~0ull;
}

// ------------------------------------------------------------------
// Kernel 1: sim = (ref . emb[:,p]) * rsqrt(sum emb[:,p]^2)  via WMMA
// One wave computes 16 pixels: D(16x16) = A(16x32) x B(32x16), K=256 in 8 steps.
// A row M=0 carries ref (branch-free: unconditional broadcast load * lane mask);
// B columns carry embedding values (f16), f32 accumulate. Column norms are
// accumulated in exact f32 in the same load loop, applied after the product
// (mathematically identical to normalizing columns first).
// ------------------------------------------------------------------
__global__ void zssam_sim_wmma_kernel(const float* __restrict__ emb,
                                      const float* __restrict__ ref,
                                      float* __restrict__ sim) {
    const int lane = threadIdx.x & 31;
    const int wave = blockIdx.x * (blockDim.x >> 5) + (threadIdx.x >> 5);
    const int n0   = wave * 16;
    const int nn   = lane & 15;   // column (pixel) within tile
    const int kg   = lane >> 4;   // K half-group per ISA 16-bit layout
    const int p    = n0 + nn;     // feature pixel this lane feeds for B
    const float arow = (nn == 0) ? 1.0f : 0.0f;  // only A-row M==0 is live

    v8f   acc = {};
    float sq  = 0.0f;

    for (int ks = 0; ks < 8; ++ks) {
        const int c0 = ks * 32;
        if (ks < 7) // gfx1250 global_prefetch of the next K-chunk
            __builtin_prefetch(&emb[(c0 + 32) * NPIX + p], 0, 3);
        v16h a, b;
#pragma unroll
        for (int i = 0; i < 16; ++i) {
            // ISA 16-bit A/B layout: lanes kg=0 hold K {0..7,16..23},
            // lanes kg=1 hold K {8..15,24..31} within each 32-chunk.
            const int k = c0 + ((i < 8) ? (kg * 8 + i) : (16 + kg * 8 + (i - 8)));
            const float v = emb[k * NPIX + p];
            sq   = fmaf(v, v, sq);
            b[i] = (_Float16)v;
            a[i] = (_Float16)(ref[k] * arow);   // branch-free A construction
        }
        acc = __builtin_amdgcn_wmma_f32_16x16x32_f16(
            /*neg_a=*/false, a, /*neg_b=*/false, b,
            /*c_mod=*/(short)0, acc, /*reuse_a=*/false, /*reuse_b=*/false);
    }
    // total column norm^2: this lane's half + the lane 16 apart (other K half)
    const float tot = sq + __shfl_xor(sq, 16, 32);
    // D layout: VGPR0, lanes 0..15 = row M=0, N=lane
    if (lane < 16) {
        sim[n0 + lane] = acc[0] * rsqrtf(tot);
    }
}

// ------------------------------------------------------------------
// Bilinear sample with half-pixel centers + edge clamp (jax.image.resize,
// method='bilinear', antialias=False semantics for upsampling)
// ------------------------------------------------------------------
__device__ __forceinline__ float zssam_bilin(const float* __restrict__ img,
                                             int h, int w, float sy, float sx) {
    const float fy = floorf(sy), fx = floorf(sx);
    const int y0 = (int)fy, x0 = (int)fx;
    const float wy = sy - fy, wx = sx - fx;
    const int y0c = min(max(y0, 0), h - 1);
    const int y1c = min(max(y0 + 1, 0), h - 1);
    const int x0c = min(max(x0, 0), w - 1);
    const int x1c = min(max(x0 + 1, 0), w - 1);
    const float v00 = img[y0c * w + x0c], v01 = img[y0c * w + x1c];
    const float v10 = img[y1c * w + x0c], v11 = img[y1c * w + x1c];
    const float top = v00 + (v01 - v00) * wx;
    const float bot = v10 + (v11 - v10) * wx;
    return top + (bot - top) * wy;
}

// ------------------------------------------------------------------
// Kernel 2: 64x64 -> 1024x1024 bilinear, keep only the 576x1024 crop
// ------------------------------------------------------------------
__global__ void zssam_upsample_kernel(const float* __restrict__ sim,
                                      float* __restrict__ u) {
    const int idx = blockIdx.x * blockDim.x + threadIdx.x;
    if (idx >= PH * PW) return;
    const int yy = idx / PW;
    const int xx = idx - yy * PW;
    const float sy = (yy + 0.5f) * (64.0f / 1024.0f) - 0.5f;
    const float sx = (xx + 0.5f) * (64.0f / 1024.0f) - 0.5f;
    u[idx] = zssam_bilin(sim, FEAT, FEAT, sy, sx);
}

// ------------------------------------------------------------------
// Kernel 3: fused second resize (576x1024 -> 2160x3840) + reductions.
// Each block's 2048 pixels span <=2 output rows; their y-taps live in u rows
// {f0, f0+1, f0+2} (f0 = floor(sy) of the block's first row). Those 3 full
// rows (12 KB) are staged into LDS via gfx1250 async-to-LDS b128 (ASYNCcnt),
// cutting the block's VMEM loads from 8192 to 768 and serving the 4-tap
// bilinear from LDS. u itself stays L2-resident (2.3 MB << 192 MB).
//  - global argmin key = (monotone_f32 << 32) | idx, minimized (tie -> min idx)
//  - per-cell argmax key = (f32_bits << 32) | ~idx, maximized (tie -> min idx);
//    valid values are > 0.65 > 0 so raw float bits order correctly.
// ------------------------------------------------------------------
__global__ void zssam_mask_reduce_kernel(const float* __restrict__ u,
                                         unsigned long long* __restrict__ cells,
                                         unsigned long long* __restrict__ gmin) {
    __shared__ __align__(16) float srows[3 * PW];      // 12 KB row cache
    __shared__ unsigned long long red[256];
    const int tid  = threadIdx.x;
    const int base = blockIdx.x * 2048;   // 8 pixels per thread, coalesced

    // ---- stage 3 clamped u rows into LDS ----
    const int ymin = base / W_OUT;
    const int f0   = (int)floorf((ymin + 0.5f) * ((float)PH / (float)H_OUT) - 0.5f);
#pragma unroll
    for (int r = 0; r < 3; ++r) {
        const int ur = min(max(f0 + r, 0), PH - 1);
        const float* gp = u + ur * PW + tid * 4;       // 256 threads x float4 = one row
#if HAVE_ASYNC_LDS
        __builtin_amdgcn_global_load_async_to_lds_b128(
            (zs_gv4p)gp, (zs_lv4p)&srows[r * PW + tid * 4], 0, 0);
#else
        *(float4*)&srows[r * PW + tid * 4] = *(const float4*)gp;
#endif
    }
#if HAVE_ASYNC_LDS
#if __has_builtin(__builtin_amdgcn_s_wait_asynccnt)
    __builtin_amdgcn_s_wait_asynccnt(0);
#else
    asm volatile("s_wait_asynccnt 0" ::: "memory");
#endif
#endif
    __syncthreads();

    unsigned long long myMin = ~0ull;
#pragma unroll
    for (int ch = 0; ch < 8; ++ch) {
        const int idx = base + ch * 256 + tid;        // < 2160*3840 by grid sizing
        const int y = idx / W_OUT;
        const int x = idx - y * W_OUT;
        const float sy = (y + 0.5f) * ((float)PH / (float)H_OUT) - 0.5f;
        const float sx = (x + 0.5f) * ((float)PW / (float)W_OUT) - 0.5f;

        // y-taps from LDS row cache: ry0 in {f0, f0+1} by construction
        const float fy = floorf(sy), fx = floorf(sx);
        const int  ry0 = (int)fy,    rx0 = (int)fx;
        const float wy = sy - fy,    wx  = sx - fx;
        const int x0c = min(max(rx0, 0), PW - 1);
        const int x1c = min(max(rx0 + 1, 0), PW - 1);
        const int r0i = ry0 - f0;                      // 0 or 1; rows pre-clamped
        const float* row0 = &srows[r0i * PW];
        const float* row1 = row0 + PW;
        const float v00 = row0[x0c], v01 = row0[x1c];
        const float v10 = row1[x0c], v11 = row1[x1c];
        const float top = v00 + (v01 - v00) * wx;
        const float bot = v10 + (v11 - v10) * wx;
        const float m   = top + (bot - top) * wy;

        const unsigned int b = __float_as_uint(m);
        const unsigned int mono = (b & 0x80000000u) ? ~b : (b | 0x80000000u);
        const unsigned long long gk =
            ((unsigned long long)mono << 32) | (unsigned int)idx;
        myMin = (gk < myMin) ? gk : myMin;

        if (m > THRESH) { // rare: sim ~ N(0,~0.1); atomics almost never fire
            const int cell = (y / CELL) * NCS + (x / CELL);
            const unsigned long long ck =
                ((unsigned long long)b << 32) | (unsigned int)(~(unsigned int)idx);
            atomicMax(&cells[cell], ck);
        }
    }
    red[tid] = myMin;
    __syncthreads();
    for (int s = 128; s > 0; s >>= 1) {
        if (tid < s) {
            const unsigned long long o = red[tid + s];
            if (o < red[tid]) red[tid] = o;
        }
        __syncthreads();
    }
    if (tid == 0) atomicMin(gmin, red[0]);
}

// ------------------------------------------------------------------
// Kernel 4: decode keys, stable sort 256 points by score desc (rank via
// O(256^2) counting in one block), emit 770 floats.
// ------------------------------------------------------------------
__global__ void zssam_finalize_kernel(const unsigned long long* __restrict__ cells,
                                      const unsigned long long* __restrict__ gmin,
                                      float* __restrict__ out) {
    __shared__ float skey[NCELLS];
    __shared__ float pts[NCELLS][3];
    const int j = threadIdx.x;

    const unsigned long long key = cells[j];
    float px = -1.0f, py = -1.0f, ps = -1.0f, sk = -INFINITY;
    if (key != 0ull) {
        const unsigned int vb  = (unsigned int)(key >> 32);
        const unsigned int idx = ~((unsigned int)(key & 0xFFFFFFFFull));
        ps = __uint_as_float(vb);
        px = (float)(idx % W_OUT);
        py = (float)(idx / W_OUT);
        sk = ps;
    }
    skey[j] = sk;
    pts[j][0] = px; pts[j][1] = py; pts[j][2] = ps;
    __syncthreads();

    int rank = 0;
    for (int i = 0; i < NCELLS; ++i) {
        const float si = skey[i];
        if (si > sk || (si == sk && i < j)) ++rank; // stable desc (-inf ties by i)
    }
    out[rank * 3 + 0] = pts[j][0];
    out[rank * 3 + 1] = pts[j][1];
    out[rank * 3 + 2] = pts[j][2];

    if (j == 0) { // background point: [col, row] of global min
        const unsigned int idx = (unsigned int)(gmin[0] & 0xFFFFFFFFull);
        out[NCELLS * 3 + 0] = (float)(idx % W_OUT);
        out[NCELLS * 3 + 1] = (float)(idx / W_OUT);
    }
}

// ------------------------------------------------------------------
extern "C" void kernel_launch(void* const* d_in, const int* in_sizes, int n_in,
                              void* d_out, int out_size, void* d_ws, size_t ws_size,
                              hipStream_t stream) {
    const float* emb = (const float*)d_in[0];  // (1,256,64,64) f32
    const float* ref = (const float*)d_in[1];  // (1,256) f32
    // d_in[2] = ori_shape {2160, 3840}; module constants in the reference.
    float* out = (float*)d_out;                // 256*3 + 2 = 770 floats
    char*  ws  = (char*)d_ws;

    float* sim = (float*)(ws + WS_SIM_OFF);
    float* u   = (float*)(ws + WS_U_OFF);
    unsigned long long* cells = (unsigned long long*)(ws + WS_CELL_OFF);
    unsigned long long* gmin  = (unsigned long long*)(ws + WS_GMIN_OFF);

    zssam_init_kernel<<<1, 256, 0, stream>>>(cells, gmin);

    // 4096 pixels / 16 per wave = 256 waves; 4 waves (128 threads) per block
    zssam_sim_wmma_kernel<<<64, 128, 0, stream>>>(emb, ref, sim);

    zssam_upsample_kernel<<<(PH * PW + 255) / 256, 256, 0, stream>>>(sim, u);

    // 2160*3840 = 8,294,400 = 4050 blocks * 256 threads * 8 pixels
    zssam_mask_reduce_kernel<<<4050, 256, 0, stream>>>(u, cells, gmin);

    zssam_finalize_kernel<<<1, NCELLS, 0, stream>>>(cells, gmin, out);
}
